// MambaEncoder_6210522710093
// MI455X (gfx1250) — compile-verified, matrix operand order checked by reading, not run on hardware
//
#include <hip/hip_runtime.h>
#include <hip/hip_bf16.h>
#include <math.h>

// ---------------------------------------------------------------------------
// Types for CDNA5 WMMA (wave32, V_WMMA_F32_16X16X32_BF16)
// ---------------------------------------------------------------------------
typedef __bf16 bf16;
typedef bf16  v16bf __attribute__((ext_vector_type(16)));
typedef bf16  v8bf  __attribute__((ext_vector_type(8)));
typedef bf16  v4bf  __attribute__((ext_vector_type(4)));
typedef float v8f   __attribute__((ext_vector_type(8)));
typedef float v4f   __attribute__((ext_vector_type(4)));

#define ACT_NONE     0
#define ACT_GELU     2
#define ACT_SOFTPLUS 3

// Force a global (address_space(1)) 16-byte load: the staging pointers are
// kept in arrays/incremented, which defeats the inferAddressSpaces pass and
// otherwise produces flat_load_b128 (which also ties up DScnt).
typedef const v4f __attribute__((address_space(1)))* gv4p;
__device__ __forceinline__ v4f load_g_v4f(const float* p) {
    return *(gv4p)(unsigned long long)p;
}

// ---------------------------------------------------------------------------
// GEMM: Y[M,N] = act( W[M,K] * X[K,N] + bias[M] ), bf16 WMMA, fp32 accum.
// REQUIRES: K % 4 == 0, N % 64 == 0.  Block = 256 thr = 8 wave32s.  Block
// tile 128(M) x 64(N); waves in 4x2 grid; each wave 32x32 = 2x2 WMMA accs;
// K stepped by 32.  Guard-free main K-loop + one guarded tail.
// ---------------------------------------------------------------------------
template <int ACT>
__global__ void __launch_bounds__(256)
k_gemm(const float* __restrict__ W, const float* __restrict__ X,
       const float* __restrict__ bias, float* __restrict__ Y,
       int M, int N, int K) {
    // 48-half row stride = 96 B -> every 8-half fragment chunk is 16B aligned
    __shared__ __align__(16) bf16 As[128][48];  // W tile,   [m][k]
    __shared__ __align__(16) bf16 Bst[64][48];  // X tile^T, [n][k]

    const int tid  = threadIdx.x;
    const int lane = tid & 31;
    const int wid  = tid >> 5;          // 0..7
    const int wm   = wid >> 1;          // 0..3  (M sub-tile, 32 rows each)
    const int wn   = wid & 1;           // 0..1  (N sub-tile, 32 cols each)
    const int half = lane >> 4;         // 0 or 1
    const int lrow = lane & 15;
    const int klo  = half * 8;          // fragment K-chunk base per ISA layout

    const int m0b = blockIdx.y * 128;
    const int n0b = blockIdx.x * 64;

    // --- loop-invariant per-thread staging descriptors ---
    const float* aptr[4];
    bf16*        asp[4];
    int          akc[4];
#pragma unroll
    for (int i = 0; i < 4; ++i) {
        const int idx = i * 256 + tid;          // 0..1023
        const int row = idx >> 3;               // 0..127
        const int kc  = (idx & 7) * 4;          // 0,4,...,28
        int gm = m0b + row;
        gm = (gm < M) ? gm : (M - 1);           // clamp: K-invariant
        aptr[i] = W + (size_t)gm * K + kc;
        asp[i]  = &As[row][kc];
        akc[i]  = kc;
    }
    const float* bptr[2];
    bf16*        bsp[2];
    int          bkk[2];
#pragma unroll
    for (int i = 0; i < 2; ++i) {
        const int idx = i * 256 + tid;          // 0..511
        const int kk  = idx >> 4;               // 0..31
        const int nc  = (idx & 15) * 4;         // 0,4,...,60
        bkk[i]  = kk;
        bptr[i] = X + (size_t)kk * N + (n0b + nc);
        bsp[i]  = &Bst[nc][kk];
    }
    const size_t bstep = (size_t)32 * N;

    v8f acc00 = {}, acc01 = {}, acc10 = {}, acc11 = {};

    const int nfull = K >> 5;
    const int ktail = K & 31;                   // multiple of 4

#define GEMM_MMA_STEP()                                                         \
    do {                                                                        \
        const int ma = wm * 32 + lrow;                                          \
        const int na = wn * 32 + lrow;                                          \
        v8bf a0l = *(const v8bf*)&As[ma][klo];                                  \
        v8bf a0h = *(const v8bf*)&As[ma][klo + 16];                             \
        v8bf a1l = *(const v8bf*)&As[ma + 16][klo];                             \
        v8bf a1h = *(const v8bf*)&As[ma + 16][klo + 16];                        \
        v8bf b0l = *(const v8bf*)&Bst[na][klo];                                 \
        v8bf b0h = *(const v8bf*)&Bst[na][klo + 16];                            \
        v8bf b1l = *(const v8bf*)&Bst[na + 16][klo];                            \
        v8bf b1h = *(const v8bf*)&Bst[na + 16][klo + 16];                       \
        v16bf a0 = __builtin_shufflevector(a0l, a0h, 0,1,2,3,4,5,6,7,8,9,10,11,12,13,14,15); \
        v16bf a1 = __builtin_shufflevector(a1l, a1h, 0,1,2,3,4,5,6,7,8,9,10,11,12,13,14,15); \
        v16bf b0 = __builtin_shufflevector(b0l, b0h, 0,1,2,3,4,5,6,7,8,9,10,11,12,13,14,15); \
        v16bf b1 = __builtin_shufflevector(b1l, b1h, 0,1,2,3,4,5,6,7,8,9,10,11,12,13,14,15); \
        acc00 = __builtin_amdgcn_wmma_f32_16x16x32_bf16(false, a0, false, b0, (short)0, acc00, false, false); \
        acc01 = __builtin_amdgcn_wmma_f32_16x16x32_bf16(false, a0, false, b1, (short)0, acc01, false, false); \
        acc10 = __builtin_amdgcn_wmma_f32_16x16x32_bf16(false, a1, false, b0, (short)0, acc10, false, false); \
        acc11 = __builtin_amdgcn_wmma_f32_16x16x32_bf16(false, a1, false, b1, (short)0, acc11, false, false); \
    } while (0)

    for (int it = 0; it < nfull; ++it) {
        // ---- guard-free staging: A 128x32, B 32x64 (transposed) ----
#pragma unroll
        for (int i = 0; i < 4; ++i) {
            v4f v = load_g_v4f(aptr[i]);
            aptr[i] += 32;
            v4bf o;
            o[0] = (bf16)v[0]; o[1] = (bf16)v[1]; o[2] = (bf16)v[2]; o[3] = (bf16)v[3];
            *(v4bf*)asp[i] = o;
        }
#pragma unroll
        for (int i = 0; i < 2; ++i) {
            v4f v = load_g_v4f(bptr[i]);
            bptr[i] += bstep;
            bf16* q = bsp[i];
            q[0] = (bf16)v[0]; q[48] = (bf16)v[1]; q[96] = (bf16)v[2]; q[144] = (bf16)v[3];
        }
        // prefetch next chunk (global_prefetch_b8; speculative, OOB is safe)
        __builtin_prefetch(aptr[0], 0, 1);
        __builtin_prefetch(bptr[0], 0, 1);
        __syncthreads();
        GEMM_MMA_STEP();
        __syncthreads();
    }

    if (ktail) {
        const int k0 = nfull * 32;
        // chunks are entirely in or entirely out (K%4==0, kc%4==0)
#pragma unroll
        for (int i = 0; i < 4; ++i) {
            const bool ok = (k0 + akc[i]) < K;
            v4f v = load_g_v4f(ok ? aptr[i] : W);
            v4bf o;
#pragma unroll
            for (int j = 0; j < 4; ++j) o[j] = ok ? (bf16)v[j] : (bf16)0.0f;
            *(v4bf*)asp[i] = o;
        }
#pragma unroll
        for (int i = 0; i < 2; ++i) {
            const bool ok = (k0 + bkk[i]) < K;
            v4f v = load_g_v4f(ok ? bptr[i] : X);
            bf16* q = bsp[i];
            q[0]   = ok ? (bf16)v[0] : (bf16)0.0f;
            q[48]  = ok ? (bf16)v[1] : (bf16)0.0f;
            q[96]  = ok ? (bf16)v[2] : (bf16)0.0f;
            q[144] = ok ? (bf16)v[3] : (bf16)0.0f;
        }
        __syncthreads();
        GEMM_MMA_STEP();
        __syncthreads();
    }
#undef GEMM_MMA_STEP

    // ---- write D: lanes 0-15: VGPR r -> M=r; lanes 16-31: M=8+r ----
    const int mbase = m0b + wm * 32 + half * 8;
    const int nbase = n0b + wn * 32 + lrow;
    const v8f* accs[4] = { &acc00, &acc01, &acc10, &acc11 };
#pragma unroll
    for (int i = 0; i < 2; ++i) {
#pragma unroll
        for (int j = 0; j < 2; ++j) {
            const v8f a = *accs[i * 2 + j];
            const int col = nbase + j * 16;
            if (col < N) {
#pragma unroll
                for (int r = 0; r < 8; ++r) {
                    const int row = mbase + i * 16 + r;
                    if (row < M) {
                        float v = a[r];
                        if (bias) v += bias[row];
                        if (ACT == ACT_GELU) {
                            v = 0.5f * v * (1.0f + erff(v * 0.70710678118654752f));
                        } else if (ACT == ACT_SOFTPLUS) {
                            v = (v > 20.0f) ? v : log1pf(__expf(v));
                        }
                        Y[(size_t)row * N + col] = v;
                    }
                }
            }
        }
    }
}

// ---------------------------------------------------------------------------
// Pad weight columns to Kp (zeros beyond Kin).  One block per row.
// ---------------------------------------------------------------------------
__global__ void k_pad_w(const float* __restrict__ w, float* __restrict__ wp,
                        int Kin, int Kp) {
    const int m = blockIdx.x;
    const int j = threadIdx.x;
    if (j < Kp) wp[(size_t)m * Kp + j] = (j < Kin) ? w[(size_t)m * Kin + j] : 0.0f;
}

// ---------------------------------------------------------------------------
// Generic 3D conv (NCDHW, OIDHW).  Grid: (ceil(H*W/256), Dout, Cout).
// Spatial dims are powers of two -> shift/mask decode, no division anywhere.
// ---------------------------------------------------------------------------
__global__ void k_conv3d(const float* __restrict__ x, const float* __restrict__ w,
                         const float* __restrict__ b, float* __restrict__ y,
                         int Cin, int Din, int Hin, int Win,
                         int Kk, int S, int P, int Hout, int Wout, int wsh) {
    const int hw  = Hout * Wout;
    const int lin = blockIdx.x * blockDim.x + threadIdx.x;
    if (lin >= hw) return;
    const int ow = lin & (Wout - 1);
    const int oh = lin >> wsh;
    const int od = blockIdx.y;
    const int co = blockIdx.z;
    const int Dout = gridDim.y;
    float acc = b[co];
    for (int ci = 0; ci < Cin; ++ci) {
        for (int kd = 0; kd < Kk; ++kd) {
            const int id = od * S - P + kd;
            if (id < 0 || id >= Din) continue;
            for (int kh = 0; kh < Kk; ++kh) {
                const int ih = oh * S - P + kh;
                if (ih < 0 || ih >= Hin) continue;
                for (int kw = 0; kw < Kk; ++kw) {
                    const int iw = ow * S - P + kw;
                    if (iw < 0 || iw >= Win) continue;
                    float xv = x[((size_t)ci * Din + id) * Hin * Win + (size_t)ih * Win + iw];
                    float wv = w[(((size_t)co * Cin + ci) * Kk + kd) * Kk * Kk + (size_t)kh * Kk + kw];
                    acc += xv * wv;
                }
            }
        }
    }
    y[((size_t)co * Dout + od) * hw + lin] = acc;
}

// ---------------------------------------------------------------------------
// Channel LayerNorm on [C, L] layout (normalize over C per position).
// ---------------------------------------------------------------------------
__global__ void k_ln_ch(const float* __restrict__ x, const float* __restrict__ w,
                        const float* __restrict__ b, float* __restrict__ y,
                        int C, int L, float eps) {
    const int l = blockIdx.x * blockDim.x + threadIdx.x;
    if (l >= L) return;
    float mu = 0.0f;
    for (int c = 0; c < C; ++c) mu += x[(size_t)c * L + l];
    mu /= (float)C;
    float var = 0.0f;
    for (int c = 0; c < C; ++c) {
        float d = x[(size_t)c * L + l] - mu;
        var += d * d;
    }
    var /= (float)C;
    float inv = rsqrtf(var + eps);
    for (int c = 0; c < C; ++c) {
        float v = (x[(size_t)c * L + l] - mu) * inv;
        y[(size_t)c * L + l] = v * w[c] + b[c];
    }
}

// ---------------------------------------------------------------------------
// Depthwise causal conv1d (k=4) + SiLU.  Grid: (ceil(L/256), D).
// ---------------------------------------------------------------------------
__global__ void k_dwconv1d_silu(const float* __restrict__ x, const float* __restrict__ w,
                                const float* __restrict__ b, float* __restrict__ y,
                                int L) {
    const int l = blockIdx.x * blockDim.x + threadIdx.x;
    const int d = blockIdx.y;
    if (l >= L) return;
    float acc = b[d];
#pragma unroll
    for (int k = 0; k < 4; ++k) {
        const int li = l - 3 + k;
        if (li >= 0) acc += w[d * 4 + k] * x[(size_t)d * L + li];
    }
    y[(size_t)d * L + l] = acc / (1.0f + __expf(-acc));
}

// ---------------------------------------------------------------------------
// Chunked parallel selective scan (linear recurrence h <- a*h + b):
//   pass1: per (d, chunk) compute transfer (P,Q): h_out = P*h_in + Q
//   pass2: per (d, n)     compose transfers over chunks -> H0 per chunk
//   pass3: per (d, chunk) rerun locally from H0 and emit gated output
// Chunk size = 64 (every L in this model is a multiple of 64).
// ---------------------------------------------------------------------------
#define SCAN_CS 64

__global__ void k_scan_pass1(const float* __restrict__ delta, const float* __restrict__ u,
                             const float* __restrict__ A_log, const float* __restrict__ Bm,
                             float* __restrict__ Pc, float* __restrict__ Qc,
                             int L, int Nc) {
    const int c = blockIdx.x * blockDim.x + threadIdx.x;
    const int d = blockIdx.y;
    if (c >= Nc) return;
    float A[16], P[16], Q[16];
#pragma unroll
    for (int n = 0; n < 16; ++n) {
        A[n] = -__expf(A_log[(size_t)d * 16 + n]);
        P[n] = 1.0f; Q[n] = 0.0f;
    }
    const int l0 = c * SCAN_CS;
    for (int i = 0; i < SCAN_CS; ++i) {
        const int l = l0 + i;
        const float dl = delta[(size_t)d * L + l];
        const float ul = u[(size_t)d * L + l];
        const float du = dl * ul;
#pragma unroll
        for (int n = 0; n < 16; ++n) {
            const float a = __expf(dl * A[n]);
            P[n] *= a;
            Q[n] = a * Q[n] + du * Bm[(size_t)n * L + l];
        }
    }
#pragma unroll
    for (int n = 0; n < 16; ++n) {
        const size_t o = (size_t)(d * 16 + n) * Nc + c;
        Pc[o] = P[n];
        Qc[o] = Q[n];
    }
}

__global__ void k_scan_pass2(const float* __restrict__ Pc, const float* __restrict__ Qc,
                             float* __restrict__ H0, int Nc, int total) {
    const int idx = blockIdx.x * blockDim.x + threadIdx.x;   // d*16+n
    if (idx >= total) return;
    const size_t base = (size_t)idx * Nc;
    float h = 0.0f;
    for (int c = 0; c < Nc; ++c) {
        H0[base + c] = h;
        h = Pc[base + c] * h + Qc[base + c];
    }
}

__global__ void k_scan_pass3(const float* __restrict__ u, const float* __restrict__ delta,
                             const float* __restrict__ A_log, const float* __restrict__ Bm,
                             const float* __restrict__ Cm, const float* __restrict__ Dv,
                             const float* __restrict__ z, const float* __restrict__ H0,
                             float* __restrict__ y, int L, int Nc) {
    const int c = blockIdx.x * blockDim.x + threadIdx.x;
    const int d = blockIdx.y;
    if (c >= Nc) return;
    float A[16], h[16];
#pragma unroll
    for (int n = 0; n < 16; ++n) {
        A[n] = -__expf(A_log[(size_t)d * 16 + n]);
        h[n] = H0[(size_t)(d * 16 + n) * Nc + c];
    }
    const float dval = Dv[d];
    const int l0 = c * SCAN_CS;
    for (int i = 0; i < SCAN_CS; ++i) {
        const int l = l0 + i;
        const float dl = delta[(size_t)d * L + l];
        const float ul = u[(size_t)d * L + l];
        const float du = dl * ul;
        float acc = 0.0f;
#pragma unroll
        for (int n = 0; n < 16; ++n) {
            h[n] = __expf(dl * A[n]) * h[n] + du * Bm[(size_t)n * L + l];
            acc += h[n] * Cm[(size_t)n * L + l];
        }
        const float yv = acc + dval * ul;
        const float zv = z[(size_t)d * L + l];
        y[(size_t)d * L + l] = yv * (zv / (1.0f + __expf(-zv)));
    }
}

// ---------------------------------------------------------------------------
// Sequence flip along L, and add-with-flip.  Grid: (ceil(L/256), C).
// ---------------------------------------------------------------------------
__global__ void k_flip(const float* __restrict__ x, float* __restrict__ y, int L) {
    const int l = blockIdx.x * blockDim.x + threadIdx.x;
    const int c = blockIdx.y;
    if (l >= L) return;
    y[(size_t)c * L + l] = x[(size_t)c * L + (L - 1 - l)];
}

__global__ void k_add_rev(float* __restrict__ out, const float* __restrict__ a,
                          const float* __restrict__ brev, int L) {
    const int l = blockIdx.x * blockDim.x + threadIdx.x;
    const int c = blockIdx.y;
    if (l >= L) return;
    out[(size_t)c * L + l] = a[(size_t)c * L + l] + brev[(size_t)c * L + (L - 1 - l)];
}

// ---------------------------------------------------------------------------
// Host orchestration
// ---------------------------------------------------------------------------
static inline void launch_gemm(const float* W, const float* X, const float* bias,
                               float* Y, int M, int N, int K, int act, hipStream_t s) {
    dim3 grid((N + 63) / 64, (M + 127) / 128);
    if (act == ACT_GELU)
        k_gemm<ACT_GELU><<<grid, 256, 0, s>>>(W, X, bias, Y, M, N, K);
    else if (act == ACT_SOFTPLUS)
        k_gemm<ACT_SOFTPLUS><<<grid, 256, 0, s>>>(W, X, bias, Y, M, N, K);
    else
        k_gemm<ACT_NONE><<<grid, 256, 0, s>>>(W, X, bias, Y, M, N, K);
}

extern "C" void kernel_launch(void* const* d_in, const int* in_sizes, int n_in,
                              void* d_out, int out_size, void* d_ws, size_t ws_size,
                              hipStream_t stream) {
    (void)in_sizes; (void)n_in; (void)out_size; (void)ws_size;

    auto in = [&](int i) -> const float* { return (const float*)d_in[i]; };

    // ---- parameter indexing (insertion-order flattening of setup_inputs) ----
    int pi = 0;
    const float* X0       = in(pi++);                       // (1,1,64,64,64)
    const float* stem_w   = in(pi++);
    const float* stem_b   = in(pi++);
    const float* stem_lnw = in(pi++);
    const float* stem_lnb = in(pi++);
    struct Down { const float *lnw, *lnb, *cw, *cb; } down[3];
    for (int i = 0; i < 3; ++i) {
        down[i].lnw = in(pi++); down[i].lnb = in(pi++);
        down[i].cw  = in(pi++); down[i].cb  = in(pi++);
    }
    struct Layer { const float* p[18]; } layers[4][2];
    for (int s = 0; s < 4; ++s)
        for (int d = 0; d < 2; ++d)
            for (int j = 0; j < 18; ++j) layers[s][d].p[j] = in(pi++);
    const float *normw[4], *normb[4];
    for (int i = 0; i < 4; ++i) { normw[i] = in(pi++); normb[i] = in(pi++); }
    struct Mlp { const float *w1, *b1, *w2, *b2; } mlp[4];
    for (int i = 0; i < 4; ++i) {
        mlp[i].w1 = in(pi++); mlp[i].b1 = in(pi++);
        mlp[i].w2 = in(pi++); mlp[i].b2 = in(pi++);
    }

    // ---- workspace arena (sized for stage 0 maxima, reused every stage) ----
    char* wsb = (char*)d_ws;
    size_t off = 0;
    auto walloc = [&](size_t nfloats) -> float* {
        float* p = (float*)(wsb + off);
        off += ((nfloats * sizeof(float) + 255) / 256) * 256;
        return p;
    };
    const size_t L0 = 32768;
    float* bx   = walloc(48 * L0);        // current activation [dim, L]
    float* bxn  = walloc(48 * L0);        // LN output / xo
    float* bxz  = walloc(192 * L0);       // in_proj output [2*d_in, L]
    float* bxzr = walloc(192 * L0);       // flipped xz
    float* bxc  = walloc(96 * L0);        // conv branch x  [d_in, L]
    float* bdbl = walloc(56 * L0);        // x_dbl [dtr+32, L]
    float* bdt  = walloc(96 * L0);        // delta [d_in, L]
    float* by1  = walloc(96 * L0);        // fwd branch y
    float* by2  = walloc(96 * L0);        // bwd branch y
    float* bys  = walloc(96 * L0);        // combined y
    float* bh   = walloc(192 * L0);       // MLP hidden [4*dim, L]
    float* bwp  = walloc(768 * 24);       // padded dt_proj weight
    const size_t scmax = (size_t)96 * 16 * (L0 / SCAN_CS);  // d_in*16*Nc max
    float* bP   = walloc(scmax);          // chunk transfer P
    float* bQ   = walloc(scmax);          // chunk transfer Q
    float* bH0  = walloc(scmax);          // chunk initial states

    const int DIMS[4] = {48, 96, 192, 384};
    const int SP[4]   = {32, 16, 8, 4};
    const int DTR[4]  = {3, 6, 12, 24};

    size_t out_off[4];
    {
        size_t o = 0;
        for (int i = 0; i < 4; ++i) {
            out_off[i] = o;
            o += (size_t)DIMS[i] * SP[i] * SP[i] * SP[i];
        }
    }

    auto cdiv = [](int a, int b) { return (a + b - 1) / b; };

    auto run_branch = [&](const float* xzp, const Layer& lp, int po,
                          int d_in, int L, int dtr, float* yout) {
        const float* conv_w = lp.p[4 + po];
        const float* conv_b = lp.p[5 + po];
        const float* xprojw = lp.p[6 + po];
        const float* dtproj = lp.p[7 + po];
        const float* dtbias = lp.p[8 + po];
        const float* Alog   = lp.p[9 + po];
        const float* Dvec   = lp.p[10 + po];
        const float* xp = xzp;
        const float* zp = xzp + (size_t)d_in * L;

        k_dwconv1d_silu<<<dim3(cdiv(L, 256), d_in), 256, 0, stream>>>(
            xp, conv_w, conv_b, bxc, L);
        // x_dbl = x_proj_w [(dtr+32) x d_in] @ xc
        launch_gemm(xprojw, bxc, nullptr, bdbl, dtr + 32, L, d_in, ACT_NONE, stream);
        // pad dt_proj weight columns to Kp % 4 == 0 (zeros multiply B rows -> no-op)
        const int Kp = (dtr + 3) & ~3;
        k_pad_w<<<d_in, 32, 0, stream>>>(dtproj, bwp, dtr, Kp);
        // delta = softplus(dt_proj_w_pad [d_in x Kp] @ x_dbl[0:Kp] + dt_bias)
        launch_gemm(bwp, bdbl, dtbias, bdt, d_in, L, Kp, ACT_SOFTPLUS, stream);
        // chunked parallel selective scan
        const float* Bm = bdbl + (size_t)dtr * L;
        const float* Cm = bdbl + (size_t)(dtr + 16) * L;
        const int Nc = L / SCAN_CS;
        k_scan_pass1<<<dim3(cdiv(Nc, 64), d_in), 64, 0, stream>>>(
            bdt, bxc, Alog, Bm, bP, bQ, L, Nc);
        k_scan_pass2<<<cdiv(d_in * 16, 256), 256, 0, stream>>>(
            bP, bQ, bH0, Nc, d_in * 16);
        k_scan_pass3<<<dim3(cdiv(Nc, 64), d_in), 64, 0, stream>>>(
            bxc, bdt, Alog, Bm, Cm, Dvec, zp, bH0, yout, L, Nc);
    };

    auto run_layer = [&](const Layer& lp, int dim, int L, int dtr) {
        const int d_in = 2 * dim;
        k_ln_ch<<<cdiv(L, 256), 256, 0, stream>>>(bx, lp.p[0], lp.p[1], bxn, dim, L, 1e-5f);
        // xz = in_proj_w [2*d_in x dim] @ xn
        launch_gemm(lp.p[2], bxn, nullptr, bxz, 2 * d_in, L, dim, ACT_NONE, stream);
        run_branch(bxz, lp, 0, d_in, L, dtr, by1);
        k_flip<<<dim3(cdiv(L, 256), 2 * d_in), 256, 0, stream>>>(bxz, bxzr, L);
        run_branch(bxzr, lp, 7, d_in, L, dtr, by2);
        k_add_rev<<<dim3(cdiv(L, 256), d_in), 256, 0, stream>>>(bys, by1, by2, L);
        // x = out_proj [dim x d_in] @ y     (overwrites bx; old x no longer used)
        launch_gemm(lp.p[3], bys, nullptr, bx, dim, L, d_in, ACT_NONE, stream);
    };

    // =========================== pipeline ===========================
    for (int s = 0; s < 4; ++s) {
        const int C = DIMS[s];
        const int R = SP[s];
        const int L = R * R * R;
        const int wsh = __builtin_ctz(R);     // log2(R), R is a power of two

        if (s == 0) {
            // stem: conv3d(1->48, k7, s2, p3) on 64^3 -> 32^3, then channel-LN
            k_conv3d<<<dim3(cdiv(R * R, 256), R, C), 256, 0, stream>>>(
                X0, stem_w, stem_b, bx, 1, 64, 64, 64, 7, 2, 3, R, R, wsh);
            k_ln_ch<<<cdiv(L, 256), 256, 0, stream>>>(bx, stem_lnw, stem_lnb, bx, C, L, 1e-6f);
        } else {
            // LN on previous-stage x, then conv3d(Cprev->C, k2, s2, p0)
            const int Cp = DIMS[s - 1];
            const int Rp = SP[s - 1];
            const int Lp = Rp * Rp * Rp;
            k_ln_ch<<<cdiv(Lp, 256), 256, 0, stream>>>(
                bx, down[s - 1].lnw, down[s - 1].lnb, bxn, Cp, Lp, 1e-6f);
            k_conv3d<<<dim3(cdiv(R * R, 256), R, C), 256, 0, stream>>>(
                bxn, down[s - 1].cw, down[s - 1].cb, bx, Cp, Rp, Rp, Rp, 2, 2, 0, R, R, wsh);
        }

        // mamba layers
        for (int d = 0; d < 2; ++d) run_layer(layers[s][d], C, L, DTR[s]);

        // output head: xo = LN(x); out_s = fc2 @ gelu(fc1 @ xo + b1) + b2
        k_ln_ch<<<cdiv(L, 256), 256, 0, stream>>>(bx, normw[s], normb[s], bxn, C, L, 1e-6f);
        launch_gemm(mlp[s].w1, bxn, mlp[s].b1, bh, 4 * C, L, C, ACT_GELU, stream);
        launch_gemm(mlp[s].w2, bh, mlp[s].b2, (float*)d_out + out_off[s], C, L, 4 * C, ACT_NONE, stream);
    }
}
